// PointRefineMaskHead_10977936408912
// MI455X (gfx1250) — matrix operand order repaired and checked by reading, not compiled, not run on hardware
//
#include <hip/hip_runtime.h>

// ---------------------------------------------------------------------------
// CDNA5 (gfx1250) bf16 WMMA types
// ---------------------------------------------------------------------------
typedef __attribute__((ext_vector_type(16))) __bf16 v16bf;
typedef __attribute__((ext_vector_type(8)))  __bf16 v8bf;
typedef __attribute__((ext_vector_type(8)))  float  v8f;

#define TM 64      // macro tile M
#define TN 128     // macro tile N
#define KC 32      // K chunk (bf16 WMMA K)
#define LDSP 40    // LDS row pitch in bf16 (80B) -> conflict-free b128 column reads

// Native hardware convert (RNE); clang lowers to v_cvt*_bf16_f32 on gfx1250.
__device__ __forceinline__ __bf16 f2bf(float f) { return (__bf16)f; }

// Fragment build + 4x WMMA for one 32K chunk from LDS buffers Asb/Bsb.
// A fragment: lane<16 holds K {0..7,16..23}, lane>=16 holds K {8..15,24..31}
// (ISA 7.12.2).  B fragment: same K split as A (lane<16 K {0..7,16..23},
// lane>=16 K {8..15,24..31}), col = lane%16.
#define WMMA_STEP(Asb, Bsb) do {                                                \
  const int lr_ = lane & 15; const int hs_ = lane >> 4;                         \
  v16bf af0_, af1_, bg0_, bg1_;                                                 \
  { int ar = wm*32 + lr_;      int k0 = hs_ ? 8 : 0;                            \
    v8bf lo = *(const v8bf*)&Asb[ar][k0]; v8bf hi = *(const v8bf*)&Asb[ar][k0+16];\
    af0_ = __builtin_shufflevector(lo,hi,0,1,2,3,4,5,6,7,8,9,10,11,12,13,14,15);}\
  { int ar = wm*32 + 16 + lr_; int k0 = hs_ ? 8 : 0;                            \
    v8bf lo = *(const v8bf*)&Asb[ar][k0]; v8bf hi = *(const v8bf*)&Asb[ar][k0+16];\
    af1_ = __builtin_shufflevector(lo,hi,0,1,2,3,4,5,6,7,8,9,10,11,12,13,14,15);}\
  { int bc = wn*32 + lr_;      int k0 = hs_ ? 8 : 0;                            \
    v8bf lo = *(const v8bf*)&Bsb[bc][k0]; v8bf hi = *(const v8bf*)&Bsb[bc][k0+16];\
    bg0_ = __builtin_shufflevector(lo,hi,0,1,2,3,4,5,6,7,8,9,10,11,12,13,14,15);}\
  { int bc = wn*32 + 16 + lr_; int k0 = hs_ ? 8 : 0;                            \
    v8bf lo = *(const v8bf*)&Bsb[bc][k0]; v8bf hi = *(const v8bf*)&Bsb[bc][k0+16];\
    bg1_ = __builtin_shufflevector(lo,hi,0,1,2,3,4,5,6,7,8,9,10,11,12,13,14,15);}\
  acc[0][0] = __builtin_amdgcn_wmma_f32_16x16x32_bf16(false, af0_, false, bg0_, (short)0, acc[0][0], false, false); \
  acc[0][1] = __builtin_amdgcn_wmma_f32_16x16x32_bf16(false, af0_, false, bg1_, (short)0, acc[0][1], false, false); \
  acc[1][0] = __builtin_amdgcn_wmma_f32_16x16x32_bf16(false, af1_, false, bg0_, (short)0, acc[1][0], false, false); \
  acc[1][1] = __builtin_amdgcn_wmma_f32_16x16x32_bf16(false, af1_, false, bg1_, (short)0, acc[1][1], false, false); \
} while (0)

// ---------------------------------------------------------------------------
// Weight transforms (fp32 -> bf16, GEMM-friendly layouts, zero padded)
// ---------------------------------------------------------------------------
__global__ void conv_w_transform(const float* __restrict__ w, __bf16* __restrict__ wT,
                                 int Cout, int Cin, int total) {
  int gt = blockIdx.x * blockDim.x + threadIdx.x;
  if (gt >= total) return;
  int o = gt % Cout; int rest = gt / Cout; int c = rest % Cin; int rs = rest / Cin;
  // src (Cout, Cin, 3, 3) -> dst [rs][c][o]
  wT[gt] = f2bf(w[((size_t)o * Cin + c) * 9 + rs]);
}

__global__ void w1x1_transform(const float* __restrict__ w, __bf16* __restrict__ wT,
                               int K, int N, int Npad, int total) {
  int gt = blockIdx.x * blockDim.x + threadIdx.x;
  if (gt >= total) return;
  int o = gt % Npad; int k = gt / Npad;
  float v = (k < K && o < N) ? w[(size_t)o * K + k] : 0.f;   // src (N, K) -> dst [k][o]
  wT[gt] = f2bf(v);
}

// ---------------------------------------------------------------------------
// conv3x3 (SAME) + bias + ReLU as implicit-im2col bf16 WMMA GEMM.
// A[m][k]: m = flat (img, y, x), k = rs*Cin + c.  D stored NCHW fp32.
// Double-buffered LDS, register-staged global prefetch, 1 barrier / chunk,
// division-free K iteration (rs, c0 advanced incrementally).
// ---------------------------------------------------------------------------
__global__ __launch_bounds__(256) void conv3x3_wmma(
    const float* __restrict__ in, const __bf16* __restrict__ wT,
    const float* __restrict__ bias, float* __restrict__ out,
    int Nimg, int Cin, int H, int Wd, int Cout) {
  const int HW = H * Wd;
  const int M  = Nimg * HW;
  const int nTN = (Cout + TN - 1) / TN;
  const int bm = blockIdx.x / nTN;
  const int bn = blockIdx.x - bm * nTN;
  const int m0 = bm * TM, n0 = bn * TN;
  const int tid  = threadIdx.x;
  const int lane = tid & 31;
  const int wm = (tid >> 5) & 1, wn = tid >> 6;
  __shared__ __align__(16) __bf16 As[2][TM][LDSP];
  __shared__ __align__(16) __bf16 Bs[2][TN][LDSP];
  v8f acc[2][2] = {};
  // A loader coords (m-fastest -> coalesced along pixels)
  const int aml = tid & 63;
  const int akb = (tid >> 6) * 8;
  const int gm  = m0 + aml;
  const bool mv = gm < M;
  const int nimg = mv ? gm / HW : 0;
  const int p    = gm - nimg * HW;
  const int py = p / Wd, px = p - (p / Wd) * Wd;
  const long pixBase = (long)nimg * Cin * HW;    // + (c)*HW + iy*Wd + ix
  // B loader coords (n-fastest -> coalesced)
  const int bnl = tid & 127;
  const int bkb = (tid >> 7) * 16;

  const int nChunks = (9 * Cin) / KC;            // Cin is a multiple of 32

  float  ra[8];
  __bf16 rb[16];

  // chunk cursor for the *next* chunk to be fetched: (rr, ss, c0), kk = flat K
  int rr = 0, ss = 0, c0 = 0, kk = 0;

  // -------- fetch chunk 0 --------
  {
    int iy = py + rr - 1, ix = px + ss - 1;
    bool ok = mv && (unsigned)iy < (unsigned)H && (unsigned)ix < (unsigned)Wd;
    long base = pixBase + (long)(c0 + akb) * HW + (long)iy * Wd + ix;
#pragma unroll
    for (int j = 0; j < 8; ++j) ra[j] = ok ? in[base + (long)j * HW] : 0.f;
#pragma unroll
    for (int j = 0; j < 16; ++j)
      rb[j] = wT[(size_t)(kk + bkb + j) * Cout + (n0 + bnl)];
  }
#pragma unroll
  for (int j = 0; j < 8; ++j)  As[0][aml][akb + j] = f2bf(ra[j]);
#pragma unroll
  for (int j = 0; j < 16; ++j) Bs[0][bnl][bkb + j] = rb[j];
  __syncthreads();
  // advance cursor to chunk 1
  c0 += KC; kk += KC;
  if (c0 == Cin) { c0 = 0; if (++ss == 3) { ss = 0; ++rr; } }

  int buf = 0;
  for (int t = 0; t < nChunks; ++t) {
    const bool more = (t + 1) < nChunks;
    if (more) {
      int iy = py + rr - 1, ix = px + ss - 1;
      bool ok = mv && (unsigned)iy < (unsigned)H && (unsigned)ix < (unsigned)Wd;
      long base = pixBase + (long)(c0 + akb) * HW + (long)iy * Wd + ix;
#pragma unroll
      for (int j = 0; j < 8; ++j) ra[j] = ok ? in[base + (long)j * HW] : 0.f;
#pragma unroll
      for (int j = 0; j < 16; ++j)
        rb[j] = wT[(size_t)(kk + bkb + j) * Cout + (n0 + bnl)];
    }
    WMMA_STEP(As[buf], Bs[buf]);
    if (more) {
#pragma unroll
      for (int j = 0; j < 8; ++j)  As[buf ^ 1][aml][akb + j] = f2bf(ra[j]);
#pragma unroll
      for (int j = 0; j < 16; ++j) Bs[buf ^ 1][bnl][bkb + j] = rb[j];
      __syncthreads();
      buf ^= 1;
      c0 += KC; kk += KC;
      if (c0 == Cin) { c0 = 0; if (++ss == 3) { ss = 0; ++rr; } }
    }
  }

  const int lr = lane & 15, hs = lane >> 4;
#pragma unroll
  for (int ni = 0; ni < 2; ++ni) {
    int col = n0 + wn * 32 + ni * 16 + lr;
    if (col >= Cout) continue;
    float bv = bias[col];
#pragma unroll
    for (int mi = 0; mi < 2; ++mi)
#pragma unroll
      for (int r = 0; r < 8; ++r) {
        int gm2 = m0 + wm * 32 + mi * 16 + r + hs * 8;
        if (gm2 < M) {
          float v = fmaxf(acc[mi][ni][r] + bv, 0.f);
          int n2 = gm2 / HW, p2 = gm2 - n2 * HW;
          out[((size_t)n2 * Cout + col) * HW + p2] = v;
        }
      }
  }
}

// ---------------------------------------------------------------------------
// Generic 1x1 GEMM:  D[M,N] = act(A[M,K] @ B[K,N] + bias)
// AM==1: A gathered from NCHW (a0=HW, a1=Cin).  AM==2: A row-major (a0=lda).
// DM==0: D scattered to NCHW (d0=HW, d1=Cout).  DM==1: D row-major (d0=ldd).
// Double-buffered LDS + register-staged prefetch, 1 barrier / chunk.
// ---------------------------------------------------------------------------
template <int AM, int DM, bool RELU>
__global__ __launch_bounds__(256) void gemm_wmma(
    const float* __restrict__ A, int a0, int a1,
    const __bf16* __restrict__ Bt, int ldb,
    const float* __restrict__ bias,
    float* __restrict__ D, int d0, int d1,
    int M, int K, int Ncols) {
  const int nTN = (Ncols + TN - 1) / TN;
  const int bm = blockIdx.x / nTN;
  const int bn = blockIdx.x - bm * nTN;
  const int m0 = bm * TM, n0 = bn * TN;
  const int tid  = threadIdx.x;
  const int lane = tid & 31;
  const int wm = (tid >> 5) & 1, wn = tid >> 6;
  __shared__ __align__(16) __bf16 As[2][TM][LDSP];
  __shared__ __align__(16) __bf16 Bs[2][TN][LDSP];
  v8f acc[2][2] = {};
  const int Kpad = (K + KC - 1) & ~(KC - 1);
  const int bnl = tid & 127;
  const int bkb = (tid >> 7) * 16;

  // A loader coords
  const int aml = tid & 63;              // AM==1
  const int akb = (tid >> 6) * 8;
  const int akl = tid & 31;              // AM==2
  const int amb = (tid >> 5) * 8;
  long aBase = 0; bool amv = false; int ap = 0;
  if (AM == 1) {
    int gmm = m0 + aml;
    amv = gmm < M;
    int nimg = amv ? gmm / a0 : 0;
    ap = gmm - nimg * a0;
    aBase = (long)nimg * a1 * a0 + ap;   // + ks*a0
  }

  float  ra[8];
  __bf16 rb[16];

  auto fetchA = [&](int kkc) {
    if (AM == 1) {
#pragma unroll
      for (int j = 0; j < 8; ++j) {
        int ks = kkc + akb + j;
        ra[j] = (amv && ks < K) ? A[aBase + (long)ks * a0] : 0.f;
      }
    } else {
      int ks = kkc + akl;
      bool kv = ks < K;
#pragma unroll
      for (int j = 0; j < 8; ++j) {
        int gmm = m0 + amb + j;
        ra[j] = (gmm < M && kv) ? A[(size_t)gmm * a0 + ks] : 0.f;
      }
    }
  };
  auto storeA = [&](int b) {
    if (AM == 1) {
#pragma unroll
      for (int j = 0; j < 8; ++j) As[b][aml][akb + j] = f2bf(ra[j]);
    } else {
#pragma unroll
      for (int j = 0; j < 8; ++j) As[b][amb + j][akl] = f2bf(ra[j]);
    }
  };

  // -------- fetch chunk 0 --------
  fetchA(0);
#pragma unroll
  for (int j = 0; j < 16; ++j) rb[j] = Bt[(size_t)(bkb + j) * ldb + (n0 + bnl)];
  storeA(0);
#pragma unroll
  for (int j = 0; j < 16; ++j) Bs[0][bnl][bkb + j] = rb[j];
  __syncthreads();

  int buf = 0;
  const int nChunks = Kpad / KC;
  for (int t = 0; t < nChunks; ++t) {
    const bool more = (t + 1) < nChunks;
    const int kkn = (t + 1) * KC;
    if (more) {
      fetchA(kkn);
#pragma unroll
      for (int j = 0; j < 16; ++j)
        rb[j] = Bt[(size_t)(kkn + bkb + j) * ldb + (n0 + bnl)];
    }
    WMMA_STEP(As[buf], Bs[buf]);
    if (more) {
      storeA(buf ^ 1);
#pragma unroll
      for (int j = 0; j < 16; ++j) Bs[buf ^ 1][bnl][bkb + j] = rb[j];
      __syncthreads();
      buf ^= 1;
    }
  }

  const int lr = lane & 15, hs = lane >> 4;
#pragma unroll
  for (int ni = 0; ni < 2; ++ni) {
    int col = n0 + wn * 32 + ni * 16 + lr;
    if (col >= Ncols) continue;
    float bv = bias[col];
#pragma unroll
    for (int mi = 0; mi < 2; ++mi)
#pragma unroll
      for (int r = 0; r < 8; ++r) {
        int gm2 = m0 + wm * 32 + mi * 16 + r + hs * 8;
        if (gm2 < M) {
          float v = acc[mi][ni][r] + bv;
          if (RELU) v = fmaxf(v, 0.f);
          if (DM == 0) {
            int n2 = gm2 / d0, p2 = gm2 - n2 * d0;
            D[((size_t)n2 * d1 + col) * d0 + p2] = v;
          } else {
            D[(size_t)gm2 * d0 + col] = v;
          }
        }
      }
  }
}

// ---------------------------------------------------------------------------
// per-ROI-label 1x1 dot:  out[n,p] = sum_c x[n,c,p] * w[label_n,c] + b[label_n]
// ---------------------------------------------------------------------------
__global__ void label_dot(const float* __restrict__ x, const float* __restrict__ wfull,
                          const float* __restrict__ bfull, const int* __restrict__ labels,
                          float* __restrict__ outMap, int Ci, int HW) {
  int n = blockIdx.y;
  int p = blockIdx.x * blockDim.x + threadIdx.x;
  if (p >= HW) return;
  int lab = labels[n];
  const float* w  = wfull + (size_t)lab * Ci;
  const float* xb = x + (size_t)n * Ci * HW + p;
  float acc = bfull[lab];
  for (int c = 0; c < Ci; ++c) acc += xb[(size_t)c * HW] * w[c];
  outMap[(size_t)n * HW + p] = acc;
}

// ---------------------------------------------------------------------------
// per-ROI top-P selection (radix select on orderable uint keys), one block/ROI
// ---------------------------------------------------------------------------
#define TOPK_CAP 3136
__global__ void topk_kernel(const float* __restrict__ detMap, int* __restrict__ pidx,
                            int HW, int P) {
  __shared__ unsigned keys[TOPK_CAP];
  __shared__ int scount;
  int n = blockIdx.x;
  const float* row = detMap + (size_t)n * HW;
  for (int i = threadIdx.x; i < HW; i += blockDim.x) {
    unsigned u = __float_as_uint(row[i]);
    keys[i] = (u & 0x80000000u) ? ~u : (u | 0x80000000u);
  }
  __syncthreads();
  unsigned X = 0;
  for (int bit = 31; bit >= 0; --bit) {
    unsigned trial = X | (1u << bit);
    if (threadIdx.x == 0) scount = 0;
    __syncthreads();
    int c = 0;
    for (int i = threadIdx.x; i < HW; i += blockDim.x) c += (keys[i] >= trial);
    atomicAdd(&scount, c);
    __syncthreads();
    if (scount >= P) X = trial;
    __syncthreads();
  }
  if (threadIdx.x == 0) scount = 0;
  __syncthreads();
  for (int i = threadIdx.x; i < HW; i += blockDim.x)
    if (keys[i] > X) { int pos = atomicAdd(&scount, 1); pidx[(size_t)n * P + pos] = i; }
  __syncthreads();
  for (int i = threadIdx.x; i < HW; i += blockDim.x)
    if (keys[i] == X) { int pos = atomicAdd(&scount, 1); if (pos < P) pidx[(size_t)n * P + pos] = i; }
}

// ---------------------------------------------------------------------------
// bilinear point sampling of sem (HWC) + cp columns into feats [M][ci+2]
// ---------------------------------------------------------------------------
__global__ void point_sample(const float* __restrict__ semHWC,
                             const float* __restrict__ instMap, const float* __restrict__ detMap,
                             const int* __restrict__ pidx, const float* __restrict__ rois,
                             float* __restrict__ featsA, float* __restrict__ featsB,
                             int Nroi, int P, int Ci, int H, int Wd, int Hs, int Ws) {
  size_t gt = (size_t)blockIdx.x * blockDim.x + threadIdx.x;
  size_t total = (size_t)Nroi * P * Ci;
  if (gt >= total) return;
  int c  = (int)(gt % Ci);
  int mp = (int)(gt / Ci);
  int p  = mp % P; int n = mp / P;
  int HWi = H * Wd;
  int idx = pidx ? pidx[(size_t)n * P + p] : p;
  float fx = ((idx % Wd) + 0.5f) / (float)Wd;
  float fy = ((idx / Wd) + 0.5f) / (float)H;
  const float* roi = rois + (size_t)n * 5;
  int   b  = (int)roi[0];
  float xi = (roi[1] + fx * (roi[3] - roi[1])) * 0.25f - 0.5f;
  float yi = (roi[2] + fy * (roi[4] - roi[2])) * 0.25f - 0.5f;
  float x0f = floorf(xi), y0f = floorf(yi);
  float dx = xi - x0f, dy = yi - y0f;
  int x0 = (int)x0f, y0 = (int)y0f;
  float v = 0.f;
#pragma unroll
  for (int q = 0; q < 4; ++q) {
    int xx = x0 + (q & 1), yy = y0 + (q >> 1);
    float wq = ((q & 1) ? dx : 1.f - dx) * ((q >> 1) ? dy : 1.f - dy);
    if (xx >= 0 && xx < Ws && yy >= 0 && yy < Hs)
      v += wq * semHWC[(((size_t)b * Hs + yy) * Ws + xx) * Ci + c];
  }
  int ld = Ci + 2;
  size_t rowo = (size_t)mp * ld;
  featsA[rowo + c] = v;
  if (c == 0) {
    float cpI = instMap[(size_t)n * HWi + idx];
    float cpD = detMap[(size_t)n * HWi + idx];
    featsA[rowo + Ci] = cpI; featsA[rowo + Ci + 1] = cpD;
    featsB[rowo + Ci] = cpI; featsB[rowo + Ci + 1] = cpD;
  }
}

// scatter pts[m][c] -> refined[n][c][idx]
__global__ void scatter_pts(const float* __restrict__ pts, const int* __restrict__ pidx,
                            float* __restrict__ refined, int Nroi, int P, int Ci, int HW) {
  size_t gt = (size_t)blockIdx.x * blockDim.x + threadIdx.x;
  size_t total = (size_t)Nroi * P * Ci;
  if (gt >= total) return;
  int c  = (int)(gt % Ci);
  int mp = (int)(gt / Ci);
  int p  = mp % P; int n = mp / P;
  int idx = pidx ? pidx[(size_t)n * P + p] : p;
  refined[((size_t)n * Ci + c) * HW + idx] = pts[(size_t)mp * Ci + c];
}

// 2x bilinear upsample (half-pixel centers, edge clamp) + ReLU, NCHW
__global__ void upsample2x(const float* __restrict__ in, float* __restrict__ out,
                           int NC, int H, int Wd) {
  int oH = 2 * H, oW = 2 * Wd;
  size_t gt = (size_t)blockIdx.x * blockDim.x + threadIdx.x;
  size_t total = (size_t)NC * oH * oW;
  if (gt >= total) return;
  int ox = (int)(gt % oW);
  size_t t2 = gt / oW;
  int oy = (int)(t2 % oH);
  int nc = (int)(t2 / oH);
  float sx = fminf(fmaxf((ox + 0.5f) * 0.5f - 0.5f, 0.f), (float)(Wd - 1));
  float sy = fminf(fmaxf((oy + 0.5f) * 0.5f - 0.5f, 0.f), (float)(H - 1));
  int x0 = (int)sx, y0 = (int)sy;
  int x1 = min(x0 + 1, Wd - 1), y1 = min(y0 + 1, H - 1);
  float dx = sx - x0, dy = sy - y0;
  const float* b = in + (size_t)nc * H * Wd;
  float v = b[(size_t)y0 * Wd + x0] * (1.f - dx) * (1.f - dy)
          + b[(size_t)y0 * Wd + x1] * dx * (1.f - dy)
          + b[(size_t)y1 * Wd + x0] * (1.f - dx) * dy
          + b[(size_t)y1 * Wd + x1] * dx * dy;
  out[gt] = fmaxf(v, 0.f);
}

// ---------------------------------------------------------------------------
// host orchestration
// ---------------------------------------------------------------------------
static inline int cdiv(long a, long b) { return (int)((a + b - 1) / b); }

extern "C" void kernel_launch(void* const* d_in, const int* in_sizes, int n_in,
                              void* d_out, int out_size, void* d_ws, size_t ws_size,
                              hipStream_t stream) {
  (void)in_sizes; (void)n_in; (void)out_size; (void)ws_size;
  const float* inst_in = (const float*)d_in[0];   // (128,256,14,14)
  const float* sem_in  = (const float*)d_in[1];   // (2,256,128,128)
  const float* rois    = (const float*)d_in[2];   // (128,5)
  const int*   labels  = (const int*)d_in[3];     // (128,)
  auto Wp = [&](int i) { return (const float*)d_in[i]; };
  // params (dict insertion order): inst_conv_ws[2]=4,5 bs=6,7 ; sem_conv_ws=8..11 bs=12..15
  // stages at 16+16*s: semw+0 semb+1 instw+2 instb+3 detw+4 detb+5 fcw+6..8 fcb+9..11
  //                    fclw+12 fclb+13 fusew+14 fuseb+15 ; final_w=64 final_b=65
  const int NROI = 128, BIMG = 2, C0 = 256, HS = 128;

  char* ws = (char*)d_ws; size_t cur = 0;
  auto alloc = [&](size_t bytes) -> void* {
    void* p = ws + cur; cur = (cur + bytes + 255) & ~(size_t)255; return p;
  };
  __bf16* wTconv = (__bf16*)alloc((size_t)9 * 256 * 256 * 2);
  __bf16* wTsm   = (__bf16*)alloc(262144);
  float* xA      = (float*)alloc((size_t)128 * 32 * 112 * 112 * 4);
  float* xB      = (float*)alloc((size_t)128 * 32 * 112 * 112 * 4);
  float* semA    = (float*)alloc((size_t)BIMG * C0 * HS * HS * 4);
  float* semB    = (float*)alloc((size_t)BIMG * C0 * HS * HS * 4);
  float* semProj = (float*)alloc((size_t)BIMG * HS * HS * 256 * 4);
  float* instMap = (float*)alloc((size_t)NROI * 3136 * 4);
  float* detMap  = (float*)alloc((size_t)NROI * 3136 * 4);
  int*   pidx    = (int*)alloc((size_t)NROI * 784 * 4);
  float* featsA  = (float*)alloc((size_t)100352 * 130 * 4);
  float* featsB  = (float*)alloc((size_t)100352 * 130 * 4);

  auto run_conv3 = [&](const float* in, int Nimg, int Cin, int H, int Wd, int Cout,
                       const float* wf, const float* bf_, float* out) {
    int tot = 9 * Cin * Cout;
    conv_w_transform<<<cdiv(tot, 256), 256, 0, stream>>>(wf, wTconv, Cout, Cin, tot);
    int M = Nimg * H * Wd;
    conv3x3_wmma<<<cdiv(M, TM) * cdiv(Cout, TN), 256, 0, stream>>>(
        in, wTconv, bf_, out, Nimg, Cin, H, Wd, Cout);
  };
  auto prep_w1 = [&](const float* wf, int K, int N) -> int {
    int Kpad = (K + 31) & ~31;
    int Npad = (N + 127) & ~127;
    int tot = Kpad * Npad;
    w1x1_transform<<<cdiv(tot, 256), 256, 0, stream>>>(wf, wTsm, K, N, Npad, tot);
    return Npad;
  };

  // instance convs: d_in0 -> xA -> xB
  run_conv3(inst_in, NROI, C0, 14, 14, C0, Wp(4), Wp(6), xA);
  run_conv3(xA,      NROI, C0, 14, 14, C0, Wp(5), Wp(7), xB);
  // semantic convs: d_in1 -> A -> B -> A -> B
  run_conv3(sem_in, BIMG, C0, HS, HS, C0, Wp(8),  Wp(12), semA);
  run_conv3(semA,   BIMG, C0, HS, HS, C0, Wp(9),  Wp(13), semB);
  run_conv3(semB,   BIMG, C0, HS, HS, C0, Wp(10), Wp(14), semA);
  run_conv3(semA,   BIMG, C0, HS, HS, C0, Wp(11), Wp(15), semB);
  const float* semFinal = semB;

  float* curx = xB;
  float* altx = xA;
  int Hc = 14, Wc = 14;
  const int CIs[3] = {256, 128, 64};
  for (int s = 0; s < 3; ++s) {
    const int base = 16 + 16 * s;
    const int ci = CIs[s], co = ci / 2;
    const int HW = Hc * Wc;
    const int P  = (HW < 784) ? HW : 784;
    const int Mpts = NROI * P;
    // semantic projection -> HWC [B*HS*HS][ci], ReLU
    {
      int Np = prep_w1(Wp(base + 0), 256, ci);
      int Msem = BIMG * HS * HS;
      gemm_wmma<1, 1, true><<<cdiv(Msem, TM) * (Np / TN), 256, 0, stream>>>(
          semFinal, HS * HS, 256, wTsm, Np, Wp(base + 1), semProj, ci, 0, Msem, 256, ci);
    }
    // label-selected inst/det prediction maps
    label_dot<<<dim3(cdiv(HW, 256), NROI), 256, 0, stream>>>(curx, Wp(base + 2), Wp(base + 3), labels, instMap, ci, HW);
    label_dot<<<dim3(cdiv(HW, 256), NROI), 256, 0, stream>>>(curx, Wp(base + 4), Wp(base + 5), labels, detMap,  ci, HW);
    const int* sp = nullptr;
    if (P < HW) { topk_kernel<<<NROI, 256, 0, stream>>>(detMap, pidx, HW, P); sp = pidx; }
    // bilinear point sampling + cp columns into feats [Mpts][ci+2]
    {
      size_t tot = (size_t)Mpts * ci;
      point_sample<<<cdiv((long)tot, 256), 256, 0, stream>>>(
          semProj, instMap, detMap, sp, rois, featsA, featsB, NROI, P, ci, Hc, Wc, HS, HS);
    }
    // FC chain (3x relu fc + concat) then fcl, all bf16 WMMA
    float* fin = featsA; float* fout = featsB;
    for (int f = 0; f < 3; ++f) {
      int Np = prep_w1(Wp(base + 6 + f), ci + 2, ci);
      gemm_wmma<2, 1, true><<<cdiv(Mpts, TM) * (Np / TN), 256, 0, stream>>>(
          fin, ci + 2, 0, wTsm, Np, Wp(base + 9 + f), fout, ci + 2, 0, Mpts, ci + 2, ci);
      float* t = fin; fin = fout; fout = t;
    }
    float* ptsbuf = fout;  // == featsA
    {
      int Np = prep_w1(Wp(base + 12), ci + 2, ci);
      gemm_wmma<2, 1, false><<<cdiv(Mpts, TM) * (Np / TN), 256, 0, stream>>>(
          fin, ci + 2, 0, wTsm, Np, Wp(base + 13), ptsbuf, ci, 0, Mpts, ci + 2, ci);
    }
    // refined = copy(x); refined[:, :, idx] = pts
    hipMemcpyAsync(altx, curx, (size_t)NROI * ci * HW * 4, hipMemcpyDeviceToDevice, stream);
    {
      size_t tot = (size_t)Mpts * ci;
      scatter_pts<<<cdiv((long)tot, 256), 256, 0, stream>>>(ptsbuf, sp, altx, NROI, P, ci, HW);
    }
    // fuse 1x1 + ReLU : refined(alt) -> fused(cur), NCHW
    {
      int Np = prep_w1(Wp(base + 14), ci, co);
      int Mf = NROI * HW;
      gemm_wmma<1, 0, true><<<cdiv(Mf, TM) * (Np / TN), 256, 0, stream>>>(
          altx, HW, ci, wTsm, Np, Wp(base + 15), curx, HW, co, Mf, ci, co);
    }
    // 2x bilinear upsample + ReLU : cur -> alt
    {
      size_t tot = (size_t)NROI * co * 4 * HW;
      upsample2x<<<cdiv((long)tot, 256), 256, 0, stream>>>(curx, altx, NROI * co, Hc, Wc);
    }
    float* t = curx; curx = altx; altx = t;
    Hc *= 2; Wc *= 2;
  }
  // final 1x1, label-selected, no ReLU -> d_out (128,1,112,112)
  label_dot<<<dim3(cdiv(Hc * Wc, 256), NROI), 256, 0, stream>>>(
      curx, Wp(64), Wp(65), labels, (float*)d_out, 32, Hc * Wc);
}
